// _BettiSketchLite_33234456936925
// MI455X (gfx1250) — matrix operand order repaired and checked by reading, not compiled, not run on hardware
//
#include <hip/hip_runtime.h>

// ---------------------------------------------------------------------------
// BettiSketchLite for MI455X (gfx1250, wave32, WMMA).
// Heavy compute = two Gram GEMMs (4096x4096x{512,256}) -> bf16 WMMA
// (v_wmma_f32_16x16x32_bf16) with fp32 accumulate.  The Gram GEMM uses a
// double-buffered GLOBAL_LOAD_ASYNC_TO_LDS_B128 pipeline (ASYNCcnt) so the
// global->LDS copy of tile t+1 overlaps the WMMAs on tile t.
// Top-k via per-row histogram selection; components via min-label + jump.
// ---------------------------------------------------------------------------

typedef __attribute__((ext_vector_type(16))) __bf16 v16bf;
typedef __attribute__((ext_vector_type(8)))  __bf16 v8bf;
typedef __attribute__((ext_vector_type(8)))  float  v8f;

static __device__ __forceinline__ int imin(int a, int b) { return a < b ? a : b; }

static __device__ __forceinline__ unsigned short f2bfbits(float f) {
  union { float f; unsigned u; } x; x.f = f;
  unsigned r = x.u + 0x7fffu + ((x.u >> 16) & 1u);   // round-to-nearest-even
  return (unsigned short)(r >> 16);
}
static __device__ __forceinline__ unsigned pk2bf(float a, float b) {
  return (unsigned)f2bfbits(a) | ((unsigned)f2bfbits(b) << 16);
}

#define BM 128
#define BN 128
#define BKK 32
#define LDT 40   // padded LDS row stride in halves (80B: conflict-free frag reads)

// ---------------------------------------------------------------------------
// Fragment loads (shared by both GEMMs).
//   A 16x32 bf16:  lanes 0-15: M=lane,    K=0..15
//                  lanes16-31: M=lane-16, K=16..31
//   B 32x16 bf16:  lanes 0-15: N=lane,    K=0..15
//                  lanes16-31: N=lane-16, K=16..31
// ---------------------------------------------------------------------------
static __device__ __forceinline__ v16bf load_afrag(const __bf16* base, int lane) {
  const __bf16* p = base + (size_t)(lane & 15) * LDT + ((lane >> 4) * 16);
  v8bf lo = *(const v8bf*)p;
  v8bf hi = *(const v8bf*)(p + 8);
  return __builtin_shufflevector(lo, hi, 0, 1, 2, 3, 4, 5, 6, 7,
                                 8, 9, 10, 11, 12, 13, 14, 15);
}
static __device__ __forceinline__ v16bf load_bfrag(const __bf16* base, int lane) {
  const __bf16* p = base + (size_t)(lane & 15) * LDT + ((lane >> 4) * 16);
  v8bf lo = *(const v8bf*)p;
  v8bf hi = *(const v8bf*)(p + 8);
  return __builtin_shufflevector(lo, hi, 0, 1, 2, 3, 4, 5, 6, 7,
                                 8, 9, 10, 11, 12, 13, 14, 15);
}

// ---------------------------------------------------------------------------
// NT GEMM, fp32 inputs: C[M,N] = A[M,K] * B[N,K]^T (convert to bf16 in LDS).
// Block tile 128x128x32; 256 threads = 8 waves; wave tile 64x32 (4x2 WMMA).
// ---------------------------------------------------------------------------
__global__ __launch_bounds__(256) void wmma_nt_gemm_f32(
    const float* __restrict__ A, const float* __restrict__ B,
    float* __restrict__ C, int M, int Nn, int K)
{
  __shared__ __bf16 sA[BM * LDT];
  __shared__ __bf16 sB[BN * LDT];

  const int tid   = threadIdx.x;
  const int lane  = tid & 31;
  const int wave  = tid >> 5;
  const int waveM = wave & 1;
  const int waveN = wave >> 1;
  const int bm = blockIdx.y * BM;
  const int bn = blockIdx.x * BN;

  v8f acc[4][2];
  {
    v8f z = {0.f, 0.f, 0.f, 0.f, 0.f, 0.f, 0.f, 0.f};
    for (int t = 0; t < 4; ++t)
      for (int u = 0; u < 2; ++u) acc[t][u] = z;
  }

  for (int k0 = 0; k0 < K; k0 += BKK) {
    // stage: 4 chunks of 4 floats per thread; load A+B first, then convert.
    for (int r = 0; r < 4; ++r) {
      int linear = (tid + r * 256) * 4;
      int row = linear >> 5, col = linear & 31;
      float4 av = *(const float4*)(A + (size_t)(bm + row) * K + k0 + col);
      float4 bv = *(const float4*)(B + (size_t)(bn + row) * K + k0 + col);
      *(unsigned*)(sA + row * LDT + col + 0) = pk2bf(av.x, av.y);
      *(unsigned*)(sA + row * LDT + col + 2) = pk2bf(av.z, av.w);
      *(unsigned*)(sB + row * LDT + col + 0) = pk2bf(bv.x, bv.y);
      *(unsigned*)(sB + row * LDT + col + 2) = pk2bf(bv.z, bv.w);
    }
    if (k0 + BKK < K) {  // global_prefetch_b8 for next tiles
      int row = tid >> 1;
      __builtin_prefetch(A + (size_t)(bm + row) * K + k0 + BKK, 0, 1);
      __builtin_prefetch(B + (size_t)(bn + row) * K + k0 + BKK, 0, 1);
    }
    __syncthreads();

    v16bf afrag[4], bfrag[2];
    for (int t = 0; t < 4; ++t)
      afrag[t] = load_afrag(sA + (size_t)(waveM * 64 + t * 16) * LDT, lane);
    for (int u = 0; u < 2; ++u)
      bfrag[u] = load_bfrag(sB + (size_t)(waveN * 32 + u * 16) * LDT, lane);

    for (int t = 0; t < 4; ++t)
      for (int u = 0; u < 2; ++u)
        acc[t][u] = __builtin_amdgcn_wmma_f32_16x16x32_bf16(
            false, afrag[t], false, bfrag[u], (short)0, acc[t][u], false, false);

    __syncthreads();
  }

  const int rhalf = (lane >> 4) * 8;
  for (int t = 0; t < 4; ++t) {
    int rowBase = bm + waveM * 64 + t * 16 + rhalf;
    for (int u = 0; u < 2; ++u) {
      int col = bn + waveN * 32 + u * 16 + (lane & 15);
      for (int v = 0; v < 8; ++v)
        C[(size_t)(rowBase + v) * Nn + col] = acc[t][u][v];
    }
  }
}

// ---------------------------------------------------------------------------
// Gram GEMM, bf16 inputs: C[N,N] = Z[N,K] * Z[N,K]^T.
// Double-buffered: tile t+1 is copied global->LDS with
// GLOBAL_LOAD_ASYNC_TO_LDS_B128 (ASYNCcnt) while tile t feeds the WMMAs.
// ---------------------------------------------------------------------------
__global__ __launch_bounds__(256) void wmma_gram_bf16(
    const __bf16* __restrict__ Z, float* __restrict__ C, int Nn, int K)
{
  __shared__ __bf16 sA[2][BM * LDT];
  __shared__ __bf16 sB[2][BN * LDT];

  const int tid   = threadIdx.x;
  const int lane  = tid & 31;
  const int wave  = tid >> 5;
  const int waveM = wave & 1;
  const int waveN = wave >> 1;
  const int bm = blockIdx.y * BM;
  const int bn = blockIdx.x * BN;

  // per-thread staging coords: 2 chunks of 8 halves (16B) each
  const int l0 = tid * 8;
  const int r0 = l0 >> 5, c0 = l0 & 31;
  const int l1 = (tid + 256) * 8;
  const int r1 = l1 >> 5, c1 = l1 & 31;

  v8f acc[4][2];
  {
    v8f z = {0.f, 0.f, 0.f, 0.f, 0.f, 0.f, 0.f, 0.f};
    for (int t = 0; t < 4; ++t)
      for (int u = 0; u < 2; ++u) acc[t][u] = z;
  }

  // issue 4 async B128 copies (A: 2 chunks, B: 2 chunks) into buffer `buf`
  auto issue_tile = [&](int buf, int k0) {
    unsigned la0 = (unsigned)(size_t)&sA[buf][r0 * LDT + c0];
    unsigned la1 = (unsigned)(size_t)&sA[buf][r1 * LDT + c1];
    unsigned lb0 = (unsigned)(size_t)&sB[buf][r0 * LDT + c0];
    unsigned lb1 = (unsigned)(size_t)&sB[buf][r1 * LDT + c1];
    unsigned long long ga0 = (unsigned long long)(Z + (size_t)(bm + r0) * K + k0 + c0);
    unsigned long long ga1 = (unsigned long long)(Z + (size_t)(bm + r1) * K + k0 + c1);
    unsigned long long gb0 = (unsigned long long)(Z + (size_t)(bn + r0) * K + k0 + c0);
    unsigned long long gb1 = (unsigned long long)(Z + (size_t)(bn + r1) * K + k0 + c1);
    asm volatile("global_load_async_to_lds_b128 %0, %1, off"
                 :: "v"(la0), "v"(ga0) : "memory");
    asm volatile("global_load_async_to_lds_b128 %0, %1, off"
                 :: "v"(la1), "v"(ga1) : "memory");
    asm volatile("global_load_async_to_lds_b128 %0, %1, off"
                 :: "v"(lb0), "v"(gb0) : "memory");
    asm volatile("global_load_async_to_lds_b128 %0, %1, off"
                 :: "v"(lb1), "v"(gb1) : "memory");
  };

  int buf = 0;
  issue_tile(0, 0);
  asm volatile("s_wait_asynccnt 0x0" ::: "memory");
  __syncthreads();

  for (int k0 = 0; k0 < K; k0 += BKK) {
    const bool more = (k0 + BKK) < K;
    if (more) issue_tile(buf ^ 1, k0 + BKK);   // overlap copy with compute

    v16bf afrag[4], bfrag[2];
    for (int t = 0; t < 4; ++t)
      afrag[t] = load_afrag(&sA[buf][(size_t)(waveM * 64 + t * 16) * LDT], lane);
    for (int u = 0; u < 2; ++u)
      bfrag[u] = load_bfrag(&sB[buf][(size_t)(waveN * 32 + u * 16) * LDT], lane);

    for (int t = 0; t < 4; ++t)
      for (int u = 0; u < 2; ++u)
        acc[t][u] = __builtin_amdgcn_wmma_f32_16x16x32_bf16(
            false, afrag[t], false, bfrag[u], (short)0, acc[t][u], false, false);

    if (more) asm volatile("s_wait_asynccnt 0x0" ::: "memory");
    __syncthreads();
    buf ^= 1;
  }

  const int rhalf = (lane >> 4) * 8;
  for (int t = 0; t < 4; ++t) {
    int rowBase = bm + waveM * 64 + t * 16 + rhalf;
    for (int u = 0; u < 2; ++u) {
      int col = bn + waveN * 32 + u * 16 + (lane & 15);
      for (int v = 0; v < 8; ++v)
        C[(size_t)(rowBase + v) * Nn + col] = acc[t][u][v];
    }
  }
}

// ---------------------------------------------------------------------------
// L2-normalize rows of Y (fp32) -> Z (bf16).  One block per row.
// ---------------------------------------------------------------------------
__global__ __launch_bounds__(256) void normalize_rows(
    const float* __restrict__ Y, __bf16* __restrict__ Z, int C)
{
  __shared__ float red[256];
  const int row = blockIdx.x;
  const float* y = Y + (size_t)row * C;
  float ss = 0.f;
  for (int c = threadIdx.x; c < C; c += 256) { float v = y[c]; ss += v * v; }
  red[threadIdx.x] = ss;
  __syncthreads();
  for (int s = 128; s > 0; s >>= 1) {
    if (threadIdx.x < s) red[threadIdx.x] += red[threadIdx.x + s];
    __syncthreads();
  }
  float scale = 1.0f / fmaxf(sqrtf(red[0]), 1e-12f);
  for (int c = threadIdx.x * 2; c < C; c += 512)   // packed bf16x2 stores
    *(unsigned*)(Z + (size_t)row * C + c) = pk2bf(y[c] * scale, y[c + 1] * scale);
}

// ---------------------------------------------------------------------------
// Per-row top-(k+1) selection (cosine sims in [-1,1]) via histogram,
// then symmetric adjacency bitset via atomicOr.  One block per row.
// ---------------------------------------------------------------------------
#define NBINS 2048
__global__ __launch_bounds__(256) void select_topk(
    const float* __restrict__ G, unsigned* __restrict__ Adj, int N, int kPlus1)
{
  __shared__ int hist[NBINS];
  __shared__ int sBin, sNeed, sCnt;
  const int i = blockIdx.x;
  const float* g = G + (size_t)i * N;
  const int nwords = N >> 5;

  for (int b = threadIdx.x; b < NBINS; b += 256) hist[b] = 0;
  if (threadIdx.x == 0) sCnt = 0;
  __syncthreads();

  for (int j = threadIdx.x; j < N; j += 256) {
    float s = fminf(fmaxf(g[j], -1.0f), 1.0f);
    int b = imin((int)((s + 1.0f) * (float)(NBINS / 2)), NBINS - 1);
    atomicAdd(&hist[b], 1);
  }
  __syncthreads();

  if (threadIdx.x == 0) {
    int cum = 0, b = NBINS - 1;
    for (; b >= 0; --b) { cum += hist[b]; if (cum >= kPlus1) break; }
    if (b < 0) b = 0;
    sBin  = b;
    sNeed = kPlus1 - (cum - hist[b]);
  }
  __syncthreads();

  const int bstar = sBin, need = sNeed;
  for (int j = threadIdx.x; j < N; j += 256) {
    float s = fminf(fmaxf(g[j], -1.0f), 1.0f);
    int b = imin((int)((s + 1.0f) * (float)(NBINS / 2)), NBINS - 1);
    bool take = false;
    if (b > bstar) take = true;
    else if (b == bstar) { int slot = atomicAdd(&sCnt, 1); take = (slot < need); }
    if (take && j != i) {
      atomicOr(&Adj[(size_t)i * nwords + (j >> 5)], 1u << (j & 31));
      atomicOr(&Adj[(size_t)j * nwords + (i >> 5)], 1u << (i & 31));
    }
  }
}

// ---------------------------------------------------------------------------
// Connected components: min-label propagation over bitset + pointer jumping.
// ---------------------------------------------------------------------------
__global__ __launch_bounds__(128) void minprop(
    const unsigned* __restrict__ Adj, const int* __restrict__ labels,
    int* __restrict__ labels2)
{
  __shared__ int red[128];
  const int i = blockIdx.x;
  const int w = threadIdx.x;                 // 128 words = 4096 bits
  unsigned bits = Adj[(size_t)i * 128 + w];
  int m = 0x7fffffff;
  while (bits) {
    int b = __builtin_ctz(bits);
    bits &= bits - 1;
    m = imin(m, labels[w * 32 + b]);
  }
  red[w] = m;
  __syncthreads();
  for (int s = 64; s > 0; s >>= 1) {
    if (w < s) red[w] = imin(red[w], red[w + s]);
    __syncthreads();
  }
  if (w == 0) labels2[i] = imin(labels[i], red[0]);
}

__global__ void jump_labels(const int* __restrict__ labels2,
                            int* __restrict__ labels, int N)
{
  int i = blockIdx.x * blockDim.x + threadIdx.x;
  if (i < N) { int l = labels2[i]; labels[i] = labels2[l]; }
}

__global__ __launch_bounds__(256) void count_comps(
    const int* __restrict__ labels, int N, int k, float* __restrict__ out)
{
  __shared__ int red[256];
  int c = 0;
  for (int i = threadIdx.x; i < N; i += 256) c += (labels[i] == i);
  red[threadIdx.x] = c;
  __syncthreads();
  for (int s = 128; s > 0; s >>= 1) {
    if (threadIdx.x < s) red[threadIdx.x] += red[threadIdx.x + s];
    __syncthreads();
  }
  if (threadIdx.x == 0) {
    int comps = red[0];
    long long b1 = (long long)N * k - N + comps;
    if (b1 < 0) b1 = 0;
    atomicAdd(&out[0], (float)comps);
    atomicAdd(&out[1], (float)b1);
  }
}

// ---------------------------------------------------------------------------
__global__ void zero_u32(unsigned* __restrict__ p, int n) {
  int i = blockIdx.x * 256 + threadIdx.x;
  if (i < n) p[i] = 0u;
}
__global__ void init_labels(int* __restrict__ l, int n) {
  int i = blockIdx.x * 256 + threadIdx.x;
  if (i < n) l[i] = i;
}
__global__ void zero_out(float* __restrict__ o) {
  if (threadIdx.x < 2) o[threadIdx.x] = 0.f;
}

// ---------------------------------------------------------------------------
extern "C" void kernel_launch(void* const* d_in, const int* in_sizes, int n_in,
                              void* d_out, int out_size, void* d_ws, size_t ws_size,
                              hipStream_t stream)
{
  (void)in_sizes; (void)n_in; (void)out_size; (void)ws_size;
  const float* feats = (const float*)d_in[0];   // [4096, 512]
  const float* Wl[2] = { (const float*)d_in[1], (const float*)d_in[2] };
  float* out = (float*)d_out;

  const int N = 4096, K = 512;
  const int Couts[2] = { 512, 256 };
  const int ks[2]    = { 409, 204 };  // max(3, int(ratio*N)) for 0.1, 0.05

  // workspace layout (~82 MB)
  char* p = (char*)d_ws;
  float*    Y       = (float*)p;    p += (size_t)N * 512 * sizeof(float);
  __bf16*   Z       = (__bf16*)p;   p += (size_t)N * 512 * sizeof(__bf16);
  float*    G       = (float*)p;    p += (size_t)N * N * sizeof(float);
  unsigned* Adj     = (unsigned*)p; p += (size_t)N * (N / 32) * sizeof(unsigned);
  int*      labels  = (int*)p;      p += (size_t)N * sizeof(int);
  int*      labels2 = (int*)p;

  zero_out<<<1, 32, 0, stream>>>(out);

  for (int l = 0; l < 2; ++l) {
    const int Co = Couts[l];
    // Y = feats @ W^T   (fp32 in, bf16 WMMA)
    wmma_nt_gemm_f32<<<dim3(Co / BN, N / BM), 256, 0, stream>>>(
        feats, Wl[l], Y, N, Co, K);
    // Z = l2norm(Y) as bf16
    normalize_rows<<<N, 256, 0, stream>>>(Y, Z, Co);
    // G = Z @ Z^T   (bf16 in, async-LDS double-buffered WMMA)
    wmma_gram_bf16<<<dim3(N / BN, N / BM), 256, 0, stream>>>(Z, G, N, Co);
    // adjacency from top-(k+1) per row (self included, then excluded)
    zero_u32<<<(N * (N / 32)) / 256, 256, 0, stream>>>(Adj, N * (N / 32));
    select_topk<<<N, 256, 0, stream>>>(G, Adj, N, ks[l] + 1);
    // components
    init_labels<<<N / 256, 256, 0, stream>>>(labels, N);
    for (int it = 0; it < 32; ++it) {
      minprop<<<N, 128, 0, stream>>>(Adj, labels, labels2);
      jump_labels<<<N / 256, 256, 0, stream>>>(labels2, labels, N);
    }
    count_comps<<<1, 256, 0, stream>>>(labels, N, ks[l], out);
  }
}
